// NeuralNetworkSimplified_65661460021769
// MI455X (gfx1250) — 8-bit matrix operand order repaired and checked by reading, compiled, not run on hardware
//
#include <hip/hip_runtime.h>
#include <stdint.h>

typedef __attribute__((ext_vector_type(16))) int   v16i;
typedef __attribute__((ext_vector_type(8)))  float v8f;

#define FP8_P1 ((uint8_t)0x38)
#define FP8_M1 ((uint8_t)0xB8)

#if defined(__HIP_DEVICE_COMPILE__) && __has_builtin(__builtin_amdgcn_global_load_async_to_lds_b128)
#define USE_ASYNC 1
typedef int v4i_n __attribute__((vector_size(16)));
typedef __attribute__((address_space(1))) v4i_n* gvec_p;
typedef __attribute__((address_space(3))) v4i_n* lvec_p;
__device__ __forceinline__ lvec_p to_lds(void* p) {
  return (lvec_p)(uint32_t)(uintptr_t)p;
}
__device__ __forceinline__ gvec_p to_glb(const void* p) {
  return (gvec_p)(uintptr_t)p;
}
#else
#define USE_ASYNC 0
#endif

#if defined(__HIP_DEVICE_COMPILE__) && __has_builtin(__builtin_amdgcn_s_wait_asynccnt)
#define WAIT_ASYNC(n) __builtin_amdgcn_s_wait_asynccnt(n)
#elif defined(__HIP_DEVICE_COMPILE__)
#define WAIT_ASYNC(n) asm volatile("s_wait_asynccnt %0" ::"i"(n) : "memory")
#else
#define WAIT_ASYNC(n)
#endif

__global__ __launch_bounds__(256) void k_bin_fp8(const float4* __restrict__ src,
                                                 uchar4* __restrict__ dst, int n4) {
  int i = blockIdx.x * blockDim.x + threadIdx.x;
  if (i >= n4) return;
  float4 v = src[i];
  uchar4 r;
  r.x = (v.x >= 0.0f) ? FP8_P1 : FP8_M1;
  r.y = (v.y >= 0.0f) ? FP8_P1 : FP8_M1;
  r.z = (v.z >= 0.0f) ? FP8_P1 : FP8_M1;
  r.w = (v.w >= 0.0f) ? FP8_P1 : FP8_M1;
  dst[i] = r;
}

#define BM 128
#define BN 128
#define BK 128
#define LDK 144

__global__ __launch_bounds__(256, 1) void k_gemm_fp8(const uint8_t* __restrict__ A,
                                                     const uint8_t* __restrict__ W,
                                                     float* __restrict__ Dm,
                                                     int M, int N, int K) {
  __shared__ __align__(16) uint8_t lA[2][BM * LDK];
  __shared__ __align__(16) uint8_t lB[2][BN * LDK];

  const int tid  = threadIdx.x;
  const int wave = tid >> 5;
  const int lane = tid & 31;

  const int m0 = blockIdx.y * BM;
  const int n0 = blockIdx.x * BN;

  const int ldRow = tid >> 1;
  const int ldOff = (tid & 1) * 64;
  const uint8_t* gA = A + (size_t)(m0 + ldRow) * (size_t)K + ldOff;
  const uint8_t* gW = W + (size_t)(n0 + ldRow) * (size_t)K + ldOff;

  v8f acc[8];
#pragma unroll
  for (int t = 0; t < 8; ++t)
    acc[t] = (v8f){0.f, 0.f, 0.f, 0.f, 0.f, 0.f, 0.f, 0.f};

  const int hf    = lane >> 4;
  const int laneR = lane & 15;

  auto compute = [&](int buf) {
    const uint8_t* aBase = &lA[buf][(16 * wave + laneR) * LDK + hf * 8];
    int2 c0 = *(const int2*)(aBase +   0);
    int2 c1 = *(const int2*)(aBase +  16);
    int2 c2 = *(const int2*)(aBase +  32);
    int2 c3 = *(const int2*)(aBase +  48);
    int2 c4 = *(const int2*)(aBase +  64);
    int2 c5 = *(const int2*)(aBase +  80);
    int2 c6 = *(const int2*)(aBase +  96);
    int2 c7 = *(const int2*)(aBase + 112);
    v16i af = {c0.x, c0.y, c1.x, c1.y, c2.x, c2.y, c3.x, c3.y,
               c4.x, c4.y, c5.x, c5.y, c6.x, c6.y, c7.x, c7.y};
#pragma unroll
    for (int t = 0; t < 8; ++t) {
      const uint8_t* bBase = &lB[buf][(t * 16 + laneR) * LDK + hf * 8];
      int2 b0 = *(const int2*)(bBase +   0);
      int2 b1 = *(const int2*)(bBase +  16);
      int2 b2 = *(const int2*)(bBase +  32);
      int2 b3 = *(const int2*)(bBase +  48);
      int2 b4 = *(const int2*)(bBase +  64);
      int2 b5 = *(const int2*)(bBase +  80);
      int2 b6 = *(const int2*)(bBase +  96);
      int2 b7 = *(const int2*)(bBase + 112);
      v16i bf = {b0.x, b0.y, b1.x, b1.y, b2.x, b2.y, b3.x, b3.y,
                 b4.x, b4.y, b5.x, b5.y, b6.x, b6.y, b7.x, b7.y};
      acc[t] = __builtin_amdgcn_wmma_f32_16x16x128_fp8_fp8(
          af, bf, (short)0, acc[t], false, false);
    }
  };

#if USE_ASYNC
  auto issue_tile = [&](int buf, int kt) {
    lvec_p dA = to_lds(&lA[buf][ldRow * LDK + ldOff]);
    lvec_p dB = to_lds(&lB[buf][ldRow * LDK + ldOff]);
    gvec_p sA = to_glb(gA + kt);
    gvec_p sB = to_glb(gW + kt);
#pragma unroll
    for (int j = 0; j < 4; ++j) {
      __builtin_amdgcn_global_load_async_to_lds_b128(sA + j, dA + j, 0, 0);
      __builtin_amdgcn_global_load_async_to_lds_b128(sB + j, dB + j, 0, 0);
    }
  };

  issue_tile(0, 0);
  for (int kt = 0; kt < K; kt += BK) {
    const int buf = (kt >> 7) & 1;
    if (kt + BK < K) {
      issue_tile(buf ^ 1, kt + BK);
      WAIT_ASYNC(8);
    } else {
      WAIT_ASYNC(0);
    }
    if (kt + 2 * BK < K) {
      __builtin_prefetch(gA + kt + 2 * BK, 0, 0);
      __builtin_prefetch(gW + kt + 2 * BK, 0, 0);
    }
    __syncthreads();
    compute(buf);
    __syncthreads();
  }
#else
  int4 ra[4], rb[4];
#pragma unroll
  for (int j = 0; j < 4; ++j) {
    ra[j] = *(const int4*)(gA + j * 16);
    rb[j] = *(const int4*)(gW + j * 16);
  }
  for (int kt = 0; kt < K; kt += BK) {
#pragma unroll
    for (int j = 0; j < 4; ++j) {
      *(int4*)&lA[0][ldRow * LDK + ldOff + j * 16] = ra[j];
      *(int4*)&lB[0][ldRow * LDK + ldOff + j * 16] = rb[j];
    }
    __syncthreads();
    if (kt + BK < K) {
#pragma unroll
      for (int j = 0; j < 4; ++j) {
        ra[j] = *(const int4*)(gA + (kt + BK) + j * 16);
        rb[j] = *(const int4*)(gW + (kt + BK) + j * 16);
      }
    }
    compute(0);
    __syncthreads();
  }
#endif

  const int row0 = m0 + 16 * wave + hf * 8;
#pragma unroll
  for (int t = 0; t < 8; ++t) {
    const int col = n0 + t * 16 + laneR;
#pragma unroll
    for (int i = 0; i < 8; ++i)
      Dm[(size_t)(row0 + i) * (size_t)N + col] = acc[t][i];
  }
}

__global__ __launch_bounds__(256) void k_colstats(const float* __restrict__ dot,
                                                  int rowsPer, int H,
                                                  double* __restrict__ sumd,
                                                  double* __restrict__ sumsqd) {
  const int col = blockIdx.x * 256 + threadIdx.x;
  const int r0  = blockIdx.y * rowsPer;
  double s = 0.0, sq = 0.0;
  for (int r = 0; r < rowsPer; ++r) {
    float v = dot[(size_t)(r0 + r) * (size_t)H + col];
    s  += (double)v;
    sq += (double)v * (double)v;
  }
  atomicAdd(&sumd[col], s);
  atomicAdd(&sumsqd[col], sq);
}

__global__ void k_coeffs(const double* __restrict__ sumd,
                         const double* __restrict__ sumsqd,
                         const float* __restrict__ gamma,
                         const float* __restrict__ beta,
                         float2* __restrict__ coef, int H, double invB) {
  int c = blockIdx.x * blockDim.x + threadIdx.x;
  if (c >= H) return;
  double mean = sumd[c] * invB;
  double var  = sumsqd[c] * invB - mean * mean;
  float inv = rsqrtf((float)var + 1e-5f);
  float s = inv * gamma[c];
  float t = beta[c] - (float)mean * s;
  coef[c] = make_float2(s, t);
}

__global__ __launch_bounds__(256) void k_bnsign(const float* __restrict__ dot,
                                                const float2* __restrict__ coef,
                                                uint8_t* __restrict__ out,
                                                int n, int colMask) {
  int i = blockIdx.x * blockDim.x + threadIdx.x;
  if (i >= n) return;
  float2 c = coef[i & colMask];
  float v = dot[i] * c.x + c.y;
  out[i] = (v >= 0.0f) ? FP8_P1 : FP8_M1;
}

__global__ __launch_bounds__(256) void k_biasout(const float* __restrict__ dot,
                                                 const float* __restrict__ b4,
                                                 float* __restrict__ out,
                                                 int n, int colMask) {
  int i = blockIdx.x * blockDim.x + threadIdx.x;
  if (i >= n) return;
  out[i] = dot[i] + ((b4[i & colMask] >= 0.0f) ? 1.0f : -1.0f);
}

extern "C" void kernel_launch(void* const* d_in, const int* in_sizes, int n_in,
                              void* d_out, int out_size, void* d_ws, size_t ws_size,
                              hipStream_t stream) {
  const float* x   = (const float*)d_in[0];
  const float* W1  = (const float*)d_in[1];
  const float* g1  = (const float*)d_in[3];
  const float* be1 = (const float*)d_in[4];
  const float* W2  = (const float*)d_in[5];
  const float* g2  = (const float*)d_in[7];
  const float* be2 = (const float*)d_in[8];
  const float* W3  = (const float*)d_in[9];
  const float* g3  = (const float*)d_in[11];
  const float* be3 = (const float*)d_in[12];
  const float* W4  = (const float*)d_in[13];
  const float* b4  = (const float*)d_in[14];
  float* out = (float*)d_out;

  constexpr int B  = 16384, D = 3072;
  constexpr int H1 = 2048, H2 = 2048, H3 = 1024, C = 512;

  uint8_t* ws = (uint8_t*)d_ws;
  size_t o = 0;
  float*   dot    = (float*)(ws + o);  o += (size_t)B * H1 * sizeof(float);
  uint8_t* sX     = ws + o;            o += (size_t)B * D;
  uint8_t* sH     = ws + o;            o += (size_t)B * H1;
  uint8_t* sW     = ws + o;            o += (size_t)D * H1;
  double*  sumd   = (double*)(ws + o); o += (size_t)H1 * sizeof(double);
  double*  sumsqd = (double*)(ws + o); o += (size_t)H1 * sizeof(double);
  float2*  coef   = (float2*)(ws + o);

  const dim3 blk(256);
  auto bin = [&](const float* s, uint8_t* d, int n) {
    int n4 = n / 4;
    k_bin_fp8<<<dim3((n4 + 255) / 256), blk, 0, stream>>>(
        (const float4*)s, (uchar4*)d, n4);
  };
  auto stats = [&](int H, const float* g, const float* be) {
    (void)hipMemsetAsync(sumd, 0, (size_t)2 * H1 * sizeof(double), stream);
    k_colstats<<<dim3(H / 256, 64), blk, 0, stream>>>(dot, B / 64, H, sumd, sumsqd);
    k_coeffs<<<dim3((H + 255) / 256), blk, 0, stream>>>(
        sumd, sumsqd, g, be, coef, H, 1.0 / (double)B);
  };

  bin(x, sX, B * D);
  bin(W1, sW, H1 * D);
  k_gemm_fp8<<<dim3(H1 / BN, B / BM), blk, 0, stream>>>(sX, sW, dot, B, H1, D);
  stats(H1, g1, be1);
  k_bnsign<<<dim3((B * H1 + 255) / 256), blk, 0, stream>>>(dot, coef, sH, B * H1, H1 - 1);

  bin(W2, sW, H2 * H1);
  k_gemm_fp8<<<dim3(H2 / BN, B / BM), blk, 0, stream>>>(sH, sW, dot, B, H2, H1);
  stats(H2, g2, be2);
  k_bnsign<<<dim3((B * H2 + 255) / 256), blk, 0, stream>>>(dot, coef, sX, B * H2, H2 - 1);

  bin(W3, sW, H3 * H2);
  k_gemm_fp8<<<dim3(H3 / BN, B / BM), blk, 0, stream>>>(sX, sW, dot, B, H3, H2);
  stats(H3, g3, be3);
  k_bnsign<<<dim3((B * H3 + 255) / 256), blk, 0, stream>>>(dot, coef, sH, B * H3, H3 - 1);

  bin(W4, sW, C * H3);
  k_gemm_fp8<<<dim3(C / BN, B / BM), blk, 0, stream>>>(sH, sW, dot, B, C, H3);
  k_biasout<<<dim3((B * C + 255) / 256), blk, 0, stream>>>(dot, b4, out, B * C, C - 1);

  (void)in_sizes; (void)n_in; (void)out_size; (void)ws_size;
}
